// LinearCausalAttention_46540265619723
// MI455X (gfx1250) — matrix operand order repaired and checked by reading, not compiled, not run on hardware
//
#include <hip/hip_runtime.h>

// ---------------------------------------------------------------------------
// Linear causal attention for MI455X (gfx1250, wave32, WMMA 16x16x32 bf16)
// B=4, T=2048, D=1024, H=16, HD=64
// Pipeline:
//   P0a: x  -> bf16                     P0b: Wq/Wk/Wv/Wo -> bf16, transposed
//   P1 : fused QKV projection GEMM (+elu+1 feature map), writes [B,H,T,HD]
//        128x128 tile / 256 thr / double-buffered LDS / async-to-LDS staging
//   P2 : chunked linear-attention scan, all matrix work via v_wmma bf16
//   P3 : output projection GEMM (+bo), fp32 result
// ---------------------------------------------------------------------------

typedef unsigned short u16;
typedef __attribute__((ext_vector_type(16))) __bf16 v16bf;
typedef __attribute__((ext_vector_type(8)))  float  v8f;

#define BATCH 4
#define SEQ   2048
#define DIM   1024
#define HEADS 16
#define HD    64
#define CHUNK 64
#define NCHUNK (SEQ / CHUNK)
#define EPS   1e-6f

// ---------------- bf16 helpers (RNE) ----------------
__device__ inline u16 f2bf(float f) {
  union { float f; unsigned u; } a; a.f = f;
  unsigned u = a.u;
  u += 0x7fffu + ((u >> 16) & 1u);
  return (u16)(u >> 16);
}
__device__ inline float bf2f(u16 h) {
  union { unsigned u; float f; } a; a.u = ((unsigned)h) << 16;
  return a.f;
}

struct B32x8 { uint4 lo, hi; };
__device__ inline v16bf frag_cast(uint4 a, uint4 b) {
  B32x8 t; t.lo = a; t.hi = b;
  return __builtin_bit_cast(v16bf, t);
}

// ---- CDNA5 async global->LDS copy (16B per lane), tracked by ASYNCcnt ----
__device__ inline void async_copy16(u16* lds_dst, const u16* gsrc) {
  unsigned loff = (unsigned)(unsigned long long)(uintptr_t)lds_dst;  // low 32b = LDS addr
  unsigned long long ga = (unsigned long long)(uintptr_t)gsrc;
  asm volatile("global_load_async_to_lds_b128 %0, %1, off"
               :: "v"(loff), "v"(ga) : "memory");
}
__device__ inline void async_wait() {
  asm volatile("s_wait_asynccnt 0x0" ::: "memory");
}

// A fragment (16x32 bf16, MxK), source row-major [row][k] in LDS, ld in u16.
// lane<16: M=lane, K = kb + {0..7, 16..23}; lane>=16: M=lane-16, K = kb + {8..15, 24..31}
__device__ inline v16bf load_a_frag(const u16* s, int ld, int row0, int kb) {
  int lane = threadIdx.x & 31;
  int l = lane & 15, hi = lane >> 4;
  const u16* r = s + (row0 + l) * ld + kb + hi * 8;
  uint4 x0 = *(const uint4*)r;
  uint4 x1 = *(const uint4*)(r + 16);
  return frag_cast(x0, x1);
}

// B fragment (32x16 bf16, KxN), source row-major [k][n] in LDS, ld in u16.
// lane<16: N=lane, K = k0 + {0..7, 16..23}; lane>=16: N=lane-16, K = k0 + {8..15, 24..31}
__device__ inline v16bf load_b_frag(const u16* s, int ld, int k0, int n0) {
  int lane = threadIdx.x & 31;
  int l = lane & 15, hi = lane >> 4;
  v16bf f;
#pragma unroll
  for (int i = 0; i < 16; ++i) {
    int k = (i < 8) ? (hi * 8 + i) : (16 + hi * 8 + (i - 8));
    f[i] = __builtin_bit_cast(__bf16, s[(k0 + k) * ld + n0 + l]);
  }
  return f;
}

// B fragment with only column N=0 populated (lanes 0 and 16), value v for every K
__device__ inline v16bf col0_frag(u16 v) {
  int lane = threadIdx.x & 31;
  u16 w = ((lane & 15) == 0) ? v : (u16)0;
  v16bf f;
#pragma unroll
  for (int i = 0; i < 16; ++i) f[i] = __builtin_bit_cast(__bf16, w);
  return f;
}

__device__ inline v8f wmma_bf16(v16bf a, v16bf b, v8f c) {
  return __builtin_amdgcn_wmma_f32_16x16x32_bf16(false, a, false, b, (short)0, c,
                                                 false, false);
}

// ---------------- P0a: fp32 -> bf16 ----------------
__global__ void k_cvt_bf16(const float* __restrict__ src, u16* __restrict__ dst, int n) {
  int i = blockIdx.x * blockDim.x + threadIdx.x;
  if (i < n) dst[i] = f2bf(src[i]);
}

// ---------------- P0b: weights -> bf16, transposed (Wt[z][k][n] = W_z[n][k]) --
__global__ void k_cvt_weights(const float* __restrict__ Wq, const float* __restrict__ Wk,
                              const float* __restrict__ Wv, const float* __restrict__ Wo,
                              u16* __restrict__ Wt) {
  int i = blockIdx.x * blockDim.x + threadIdx.x;
  if (i >= 4 * DIM * DIM) return;
  int z = i >> 20;
  int rem = i & (DIM * DIM - 1);
  int k = rem >> 10, n = rem & (DIM - 1);
  const float* W = (z == 0) ? Wq : (z == 1) ? Wk : (z == 2) ? Wv : Wo;
  Wt[i] = f2bf(W[n * DIM + k]);
}

// ---------------- P1: QKV projection GEMM + phi, write [B,H,T,HD] ------------
// 256 thr = 8 waves in 2(M) x 4(N) grid; tile 128x128; K-step 32, double-buffered.
// Wave owns 64(M) x 32(N): 8 C-tiles, 8 wmma per K-stage.
__global__ __launch_bounds__(256) void k_qkv_gemm(
    const u16* __restrict__ xb, const u16* __restrict__ Wt3,
    const float* __restrict__ bq, const float* __restrict__ bk,
    const float* __restrict__ bv,
    u16* __restrict__ qp, u16* __restrict__ kp, u16* __restrict__ vp) {
  const int z = blockIdx.z;
  const int m0 = blockIdx.x * 128, n0 = blockIdx.y * 128;
  const u16* Wt = Wt3 + (size_t)z * DIM * DIM;
  const float* bias = (z == 0) ? bq : (z == 1) ? bk : bv;
  u16* dst = (z == 0) ? qp : (z == 1) ? kp : vp;

  __shared__ __align__(16) u16 As[2][128 * 40];  // 128 x 32 (+8 pad)
  __shared__ __align__(16) u16 Bs[2][32 * 136];  // 32 x 128 (+8 pad)

  const int tid = threadIdx.x;
  const int wave = tid >> 5, lane = tid & 31;
  const int l = lane & 15, hi = lane >> 4;
  const int wm = wave >> 2, wn = wave & 3;   // 2 x 4 wave grid

  v8f acc[8];
#pragma unroll
  for (int i = 0; i < 8; ++i)
#pragma unroll
    for (int r = 0; r < 8; ++r) acc[i][r] = 0.0f;

  auto stage = [&](int buf, int kt) {
    const int K0 = kt * 32;
#pragma unroll
    for (int i = 0; i < 2; ++i) {          // A: 128x32 = 512 x 16B
      int v = tid + i * 256;
      int r = v >> 2, c = (v & 3) * 8;
      async_copy16(&As[buf][r * 40 + c], &xb[(size_t)(m0 + r) * DIM + K0 + c]);
    }
#pragma unroll
    for (int i = 0; i < 2; ++i) {          // B: 32x128 = 512 x 16B
      int v = tid + i * 256;
      int r = v >> 4, c = (v & 15) * 8;
      async_copy16(&Bs[buf][r * 136 + c], &Wt[(size_t)(K0 + r) * DIM + n0 + c]);
    }
  };

  stage(0, 0);
  async_wait();
  __syncthreads();

  for (int kt = 0; kt < DIM / 32; ++kt) {
    const int buf = kt & 1;
    if (kt + 1 < DIM / 32) stage(buf ^ 1, kt + 1);  // overlap with compute
    v16bf bf0 = load_b_frag(&Bs[buf][0], 136, 0, wn * 32);
    v16bf bf1 = load_b_frag(&Bs[buf][0], 136, 0, wn * 32 + 16);
#pragma unroll
    for (int mi = 0; mi < 4; ++mi) {
      v16bf af = load_a_frag(&As[buf][0], 40, wm * 64 + mi * 16, 0);
      acc[mi * 2]     = wmma_bf16(af, bf0, acc[mi * 2]);
      acc[mi * 2 + 1] = wmma_bf16(af, bf1, acc[mi * 2 + 1]);
    }
    async_wait();
    __syncthreads();
  }

#pragma unroll
  for (int mi = 0; mi < 4; ++mi)
#pragma unroll
    for (int ni = 0; ni < 2; ++ni)
#pragma unroll
      for (int r = 0; r < 8; ++r) {
        int m = m0 + wm * 64 + mi * 16 + r + hi * 8;
        int n = n0 + wn * 32 + ni * 16 + l;
        float val = acc[mi * 2 + ni][r] + bias[n];
        if (z < 2) val = (val > 0.0f) ? (val + 1.0f) : __expf(val);  // elu(x)+1
        int b = m >> 11, t = m & (SEQ - 1);
        int h = n >> 6, hd = n & (HD - 1);
        dst[(((size_t)b * HEADS + h) * SEQ + t) * HD + hd] = f2bf(val);
      }
}

// ---------------- P2: chunked linear attention scan --------------------------
// grid = (B*H, 2): each WG owns one (b,h) stream and half the HD output columns.
// block = 128 (4 waves); wave w owns chunk rows [16w,16w+16) and state rows
// d in [16w,16w+16). fp32 KV state lives in WMMA accumulators across chunks.
__global__ __launch_bounds__(128) void k_lin_attn(
    const u16* __restrict__ qp, const u16* __restrict__ kp,
    const u16* __restrict__ vp, u16* __restrict__ attn) {
  const int bh = blockIdx.x;           // b*16 + h
  const int esel = blockIdx.y;         // which 32-wide slice of HD
  const int b = bh >> 4, h = bh & 15;
  const size_t base = (size_t)bh * SEQ * HD;

  __shared__ __align__(16) u16 Qc[64 * 72];   // [t][d]
  __shared__ __align__(16) u16 Kt[64 * 72];   // [d][t]  (transposed K chunk)
  __shared__ __align__(16) u16 Vc[64 * 40];   // [t][e]  (32 e-cols)
  __shared__ __align__(16) u16 KVp[64 * 40];  // [d][e]  bf16 snapshot of state
  __shared__ __align__(16) u16 Sm[64 * 72];   // [i][j]  masked scores
  __shared__ float ksum[64];

  const int tid = threadIdx.x;
  const int wave = tid >> 5, lane = tid & 31;
  const int l = lane & 15, hi = lane >> 4;

  v8f kv[2];                                   // state rows [16w,16w+16) x 32 e
#pragma unroll
  for (int e = 0; e < 2; ++e)
#pragma unroll
    for (int r = 0; r < 8; ++r) kv[e][r] = 0.0f;
  if (tid < 64) ksum[tid] = 0.0f;

  const v16bf bones = col0_frag((u16)0x3F80);  // column of 1.0

  for (int c = 0; c < NCHUNK; ++c) {
    const int t0 = c * CHUNK;
    // ---- stage chunk into LDS (Q/V via async DMA path, K via transpose) ----
    const u16* qsrc = qp + base + (size_t)t0 * HD;
    const u16* ksrc = kp + base + (size_t)t0 * HD;
    const u16* vsrc = vp + base + (size_t)t0 * HD;
#pragma unroll
    for (int i = 0; i < 4; ++i) {                    // Qc 64x64 = 512 x 16B
      int v = tid + i * 128;
      int r = v >> 3, col = (v & 7) * 8;
      async_copy16(&Qc[r * 72 + col], &qsrc[r * HD + col]);
    }
#pragma unroll
    for (int i = 0; i < 2; ++i) {                    // Vc 64x32 = 256 x 16B
      int v = tid + i * 128;
      int r = v >> 2, col = (v & 3) * 8;
      async_copy16(&Vc[r * 40 + col], &vsrc[r * HD + esel * 32 + col]);
    }
#pragma unroll
    for (int i = 0; i < 32; ++i) {                   // Kt transpose 64x64
      int v = tid + i * 128;
      int j = v >> 6, d = v & 63;
      Kt[d * 72 + j] = ksrc[j * HD + d];
    }
    async_wait();
    __syncthreads();

    // ---- snapshot state (pre-chunk) to LDS; read ksum (pre-chunk) ----
#pragma unroll
    for (int e = 0; e < 2; ++e)
#pragma unroll
      for (int r = 0; r < 8; ++r) {
        int d = wave * 16 + r + hi * 8;
        KVp[d * 40 + e * 16 + l] = f2bf(kv[e][r]);
      }
    v16bf bden[2];
#pragma unroll
    for (int kb = 0; kb < 2; ++kb)
#pragma unroll
      for (int i = 0; i < 16; ++i) {
        int k = kb * 32 + ((i < 8) ? (hi * 8 + i) : (16 + hi * 8 + (i - 8)));
        bden[kb][i] = __builtin_bit_cast(__bf16, (l == 0) ? f2bf(ksum[k]) : (u16)0);
      }
    __syncthreads();

    // ---- S = Q Kt (16x64 per wave), den += Q . ksum ----
    v16bf aq[2];
    aq[0] = load_a_frag(Qc, 72, wave * 16, 0);
    aq[1] = load_a_frag(Qc, 72, wave * 16, 32);
    v8f den;
#pragma unroll
    for (int r = 0; r < 8; ++r) den[r] = 0.0f;
    den = wmma_bf16(aq[0], bden[0], den);
    den = wmma_bf16(aq[1], bden[1], den);

    v8f s[4];
#pragma unroll
    for (int jb = 0; jb < 4; ++jb) {
#pragma unroll
      for (int r = 0; r < 8; ++r) s[jb][r] = 0.0f;
      s[jb] = wmma_bf16(aq[0], load_b_frag(Kt, 72, 0, jb * 16), s[jb]);
      s[jb] = wmma_bf16(aq[1], load_b_frag(Kt, 72, 32, jb * 16), s[jb]);
    }
    // causal mask (j <= i) and round-trip to A-layout via wave-private LDS rows
#pragma unroll
    for (int jb = 0; jb < 4; ++jb)
#pragma unroll
      for (int r = 0; r < 8; ++r) {
        int i = wave * 16 + r + hi * 8;
        int j = jb * 16 + l;
        float ms = (j <= i) ? s[jb][r] : 0.0f;
        Sm[i * 72 + j] = f2bf(ms);
      }
    v16bf as2[2];
    as2[0] = load_a_frag(Sm, 72, wave * 16, 0);
    as2[1] = load_a_frag(Sm, 72, wave * 16, 32);
    den = wmma_bf16(as2[0], bones, den);   // += rowsum(masked S)
    den = wmma_bf16(as2[1], bones, den);

    // ---- O = Sm.V (intra) + Q.KVprev (inter), 16x32 per wave ----
    v8f o[2];
#pragma unroll
    for (int e = 0; e < 2; ++e) {
#pragma unroll
      for (int r = 0; r < 8; ++r) o[e][r] = 0.0f;
      o[e] = wmma_bf16(as2[0], load_b_frag(Vc, 40, 0, e * 16), o[e]);
      o[e] = wmma_bf16(as2[1], load_b_frag(Vc, 40, 32, e * 16), o[e]);
      o[e] = wmma_bf16(aq[0], load_b_frag(KVp, 40, 0, e * 16), o[e]);
      o[e] = wmma_bf16(aq[1], load_b_frag(KVp, 40, 32, e * 16), o[e]);
    }

    // ---- normalize and store to attn [B,T,D] ----
#pragma unroll
    for (int r = 0; r < 8; ++r) {
      float dr = __shfl(den[r], 0, 16) + EPS;   // N=0 column -> lanes 0/16
      int tg = t0 + wave * 16 + r + hi * 8;
      size_t orow = ((size_t)b * SEQ + tg) * DIM + h * HD + esel * 32;
      attn[orow + l]      = f2bf(o[0][r] / dr);
      attn[orow + 16 + l] = f2bf(o[1][r] / dr);
    }

    // ---- state update: KV += K^T V, ksum += sum_j k_j ----
    v16bf ak[2];
    ak[0] = load_a_frag(Kt, 72, wave * 16, 0);
    ak[1] = load_a_frag(Kt, 72, wave * 16, 32);
#pragma unroll
    for (int e = 0; e < 2; ++e) {
      kv[e] = wmma_bf16(ak[0], load_b_frag(Vc, 40, 0, e * 16), kv[e]);
      kv[e] = wmma_bf16(ak[1], load_b_frag(Vc, 40, 32, e * 16), kv[e]);
    }
    if (tid < 64) {
      float ssum = 0.0f;
#pragma unroll
      for (int j = 0; j < 64; ++j) ssum += bf2f(Kt[tid * 72 + j]);
      ksum[tid] += ssum;
    }
    __syncthreads();
  }
}

// ---------------- P3: output projection GEMM (fp32 result) -------------------
__global__ __launch_bounds__(256) void k_out_gemm(
    const u16* __restrict__ attn, const u16* __restrict__ WoT,
    const float* __restrict__ bo, float* __restrict__ out) {
  const int m0 = blockIdx.x * 128, n0 = blockIdx.y * 128;
  __shared__ __align__(16) u16 As[2][128 * 40];
  __shared__ __align__(16) u16 Bs[2][32 * 136];
  const int tid = threadIdx.x;
  const int wave = tid >> 5, lane = tid & 31;
  const int l = lane & 15, hi = lane >> 4;
  const int wm = wave >> 2, wn = wave & 3;

  v8f acc[8];
#pragma unroll
  for (int i = 0; i < 8; ++i)
#pragma unroll
    for (int r = 0; r < 8; ++r) acc[i][r] = 0.0f;

  auto stage = [&](int buf, int kt) {
    const int K0 = kt * 32;
#pragma unroll
    for (int i = 0; i < 2; ++i) {
      int v = tid + i * 256;
      int r = v >> 2, c = (v & 3) * 8;
      async_copy16(&As[buf][r * 40 + c], &attn[(size_t)(m0 + r) * DIM + K0 + c]);
    }
#pragma unroll
    for (int i = 0; i < 2; ++i) {
      int v = tid + i * 256;
      int r = v >> 4, c = (v & 15) * 8;
      async_copy16(&Bs[buf][r * 136 + c], &WoT[(size_t)(K0 + r) * DIM + n0 + c]);
    }
  };

  stage(0, 0);
  async_wait();
  __syncthreads();

  for (int kt = 0; kt < DIM / 32; ++kt) {
    const int buf = kt & 1;
    if (kt + 1 < DIM / 32) stage(buf ^ 1, kt + 1);
    v16bf bf0 = load_b_frag(&Bs[buf][0], 136, 0, wn * 32);
    v16bf bf1 = load_b_frag(&Bs[buf][0], 136, 0, wn * 32 + 16);
#pragma unroll
    for (int mi = 0; mi < 4; ++mi) {
      v16bf af = load_a_frag(&As[buf][0], 40, wm * 64 + mi * 16, 0);
      acc[mi * 2]     = wmma_bf16(af, bf0, acc[mi * 2]);
      acc[mi * 2 + 1] = wmma_bf16(af, bf1, acc[mi * 2 + 1]);
    }
    async_wait();
    __syncthreads();
  }

#pragma unroll
  for (int mi = 0; mi < 4; ++mi)
#pragma unroll
    for (int ni = 0; ni < 2; ++ni)
#pragma unroll
      for (int r = 0; r < 8; ++r) {
        int m = m0 + wm * 64 + mi * 16 + r + hi * 8;
        int n = n0 + wn * 32 + ni * 16 + l;
        out[(size_t)m * DIM + n] = acc[mi * 2 + ni][r] + bo[n];
      }
}

// ---------------------------------------------------------------------------
extern "C" void kernel_launch(void* const* d_in, const int* in_sizes, int n_in,
                              void* d_out, int out_size, void* d_ws, size_t ws_size,
                              hipStream_t stream) {
  (void)in_sizes; (void)n_in; (void)out_size; (void)ws_size;
  const float* x  = (const float*)d_in[0];
  const float* Wq = (const float*)d_in[1];
  const float* bq = (const float*)d_in[2];
  const float* Wk = (const float*)d_in[3];
  const float* bk = (const float*)d_in[4];
  const float* Wv = (const float*)d_in[5];
  const float* bv = (const float*)d_in[6];
  const float* Wo = (const float*)d_in[7];
  const float* bo = (const float*)d_in[8];
  float* out = (float*)d_out;

  const size_t MB = 1ull << 20;
  char* ws = (char*)d_ws;
  u16* xb   = (u16*)(ws + 0);        // 16 MB: x in bf16 [8192][1024]
  u16* Wt   = (u16*)(ws + 16 * MB);  //  8 MB: 4x transposed bf16 weights
  u16* qp   = (u16*)(ws + 24 * MB);  // 16 MB: phi(q) [B,H,T,HD]
  u16* kp   = (u16*)(ws + 40 * MB);  // 16 MB: phi(k)
  u16* vp   = (u16*)(ws + 56 * MB);  // 16 MB: v
  u16* attn = (u16*)(ws + 72 * MB);  // 16 MB: attention output [B,T,D]

  const int NX = BATCH * SEQ * DIM;  // 8388608
  k_cvt_bf16<<<(NX + 255) / 256, 256, 0, stream>>>(x, xb, NX);
  k_cvt_weights<<<(4 * DIM * DIM + 255) / 256, 256, 0, stream>>>(Wq, Wk, Wv, Wo, Wt);
  k_qkv_gemm<<<dim3(BATCH * SEQ / 128, DIM / 128, 3), 256, 0, stream>>>(
      xb, Wt, bq, bk, bv, qp, kp, vp);
  k_lin_attn<<<dim3(BATCH * HEADS, 2), 128, 0, stream>>>(qp, kp, vp, attn);
  k_out_gemm<<<dim3(BATCH * SEQ / 128, DIM / 128), 256, 0, stream>>>(
      attn, Wt + 3ull * DIM * DIM, bo, out);
}
